// GAT_46445776339723
// MI455X (gfx1250) — compile-verified, matrix operand order checked by reading, not run on hardware
//
#include <hip/hip_runtime.h>
#include <math.h>

namespace {

constexpr float NEG_ATT = 0.2f;
constexpr float NEG_OUT = 0.01f;
constexpr float FMIN    = -3.402823466e38f;

typedef _Float16 half8  __attribute__((ext_vector_type(8)));
typedef _Float16 half16 __attribute__((ext_vector_type(16)));
typedef float    floatx8 __attribute__((ext_vector_type(8)));

__device__ __forceinline__ float leaky(float v, float s) { return v > 0.0f ? v : s * v; }

// Float atomic max via ordered integer representation (works for any sign mix,
// provided the location is initialized to -FLT_MAX).
__device__ __forceinline__ void atomicMaxF(float* addr, float val) {
  if (val >= 0.0f)
    atomicMax((int*)addr, __float_as_int(val));
  else
    atomicMin((unsigned int*)addr, __float_as_uint(val));
}

// Edge e in [0,E) comes from edge_index; e in [E, E+N) is a self loop.
__device__ __forceinline__ void edge_nodes(const int* __restrict__ ei, int e, int E,
                                           int& s, int& d) {
  if (e < E) { s = ei[e]; d = ei[E + e]; } else { s = e - E; d = s; }
}

// ---------------------------------------------------------------- prep kernels

__global__ void k_f32_to_f16(const float* __restrict__ in, _Float16* __restrict__ out, int n) {
  int i = blockIdx.x * blockDim.x + threadIdx.x;
  if (i < n) out[i] = (_Float16)in[i];
}

// W[K][N] (f32, row major) -> Wt[N][K] (f16, row major)
__global__ void k_transpose_f16(const float* __restrict__ W, _Float16* __restrict__ Wt,
                                int K, int N) {
  int i = blockIdx.x * blockDim.x + threadIdx.x;
  if (i >= K * N) return;
  int n = i / K, k = i - n * K;
  Wt[i] = (_Float16)W[k * N + n];
}

__global__ void k_fill_f32(float* __restrict__ p, float v, int n) {
  int i = blockIdx.x * blockDim.x + threadIdx.x;
  if (i < n) p[i] = v;
}

// In-place reciprocal of the softmax denominators: den <- 1/(den + eps).
// Moves the divide out of the per-(edge,feature) scatter (176M ops -> 0.4M).
__global__ void k_recip(float* __restrict__ p, int n) {
  int i = blockIdx.x * blockDim.x + threadIdx.x;
  if (i < n) p[i] = 1.0f / (p[i] + 1e-16f);
}

// -------------------------------------------------------------- WMMA GEMM
// C[M][N] = A[M][K] * Bt[N][K]^T   (A,Bt f16; C f32)
// One wave32 per 16x16 output tile, K stepped by 32 via v_wmma_f32_16x16x32_f16.
// Fragment layout per CDNA5 ISA 7.12.2:
//   A: lane[3:0]=row, koff = lane[4]*8; elements 0..7 -> K=koff..koff+7,
//      elements 8..15 -> K=koff+16..koff+23.   (B symmetric with lane -> column)
//   D: VGPR r -> M = r + 8*lane[4], N = lane[3:0].
__global__ void k_gemm_wmma(const _Float16* __restrict__ A,
                            const _Float16* __restrict__ Bt,
                            float* __restrict__ C,
                            int M, int N, int K) {
  const int lane = threadIdx.x & 31;
  const int waveInBlock = threadIdx.x >> 5;
  const int tilesN = N >> 4;
  const int tilesM = (M + 15) >> 4;
  int w = blockIdx.x * (blockDim.x >> 5) + waveInBlock;
  int tile_m = w / tilesN;
  int tile_n = w - tile_m * tilesN;
  if (tile_m >= tilesM) return;

  const int r15  = lane & 15;
  const int koff = (lane >> 4) << 3;   // 0 or 8
  int arow = tile_m * 16 + r15;
  if (arow >= M) arow = M - 1;         // clamped read; store is guarded
  const _Float16* ap = A  + (size_t)arow * K + koff;
  const _Float16* bp = Bt + (size_t)(tile_n * 16 + r15) * K + koff;

  floatx8 acc = {};
  for (int k0 = 0; k0 < K; k0 += 32) {
    half8 a_lo = *(const half8*)(ap + k0);
    half8 a_hi = *(const half8*)(ap + k0 + 16);
    half8 b_lo = *(const half8*)(bp + k0);
    half8 b_hi = *(const half8*)(bp + k0 + 16);
    half16 av, bv;
#pragma unroll
    for (int i = 0; i < 8; ++i) {
      av[i] = a_lo[i]; av[i + 8] = a_hi[i];
      bv[i] = b_lo[i]; bv[i + 8] = b_hi[i];
    }
    acc = __builtin_amdgcn_wmma_f32_16x16x32_f16(
        /*neg_a=*/false, av, /*neg_b=*/false, bv,
        /*c_mod=*/(short)0, acc, /*reuse_a=*/false, /*reuse_b=*/false);
  }

  const int colN  = tile_n * 16 + r15;
  const int rbase = tile_m * 16 + ((lane >> 4) << 3);
#pragma unroll
  for (int r = 0; r < 8; ++r) {
    int orow = rbase + r;
    if (orow < M) C[(size_t)orow * N + colN] = acc[r];
  }
}

// -------------------------------------------------------------- edge kernels

// Pass 1: per-(edge,head) GATv2 score a^T leakyrelu(xl[src]+xr[dst]); atomic max per dst.
template <int H, int F>
__global__ void k_edge_score(const float* __restrict__ xl, const float* __restrict__ xr,
                             const float* __restrict__ att, const int* __restrict__ ei,
                             int E, int ET, float* __restrict__ alpha, float* __restrict__ m) {
  int t = blockIdx.x * blockDim.x + threadIdx.x;
  if (t >= ET * H) return;
  int e = t / H, h = t - e * H;
  int s, d; edge_nodes(ei, e, E, s, d);
  const float4* pl = (const float4*)(xl + (size_t)s * (H * F) + h * F);
  const float4* pr = (const float4*)(xr + (size_t)d * (H * F) + h * F);
  const float4* pa = (const float4*)(att + h * F);
  float sum = 0.0f;
#pragma unroll
  for (int q = 0; q < F / 4; ++q) {
    float4 a = pl[q], b = pr[q], c = pa[q];
    sum += c.x * leaky(a.x + b.x, NEG_ATT);
    sum += c.y * leaky(a.y + b.y, NEG_ATT);
    sum += c.z * leaky(a.z + b.z, NEG_ATT);
    sum += c.w * leaky(a.w + b.w, NEG_ATT);
  }
  alpha[t] = sum;
  atomicMaxF(&m[d * H + h], sum);
}

// Pass 2: exp(score - max), accumulate softmax denominator per dst.
template <int H>
__global__ void k_edge_expsum(const int* __restrict__ ei, int E, int ET,
                              float* __restrict__ alpha, const float* __restrict__ m,
                              float* __restrict__ den) {
  int t = blockIdx.x * blockDim.x + threadIdx.x;
  if (t >= ET * H) return;
  int e = t / H, h = t - e * H;
  int s, d; edge_nodes(ei, e, E, s, d); (void)s;
  float a = __expf(alpha[t] - m[d * H + h]);
  alpha[t] = a;
  atomicAdd(&den[d * H + h], a);
}

// Pass 3 (layer 1, H=8,F=32): one block of 256 threads per edge; thread = concat
// feature. `inv` holds the precomputed 1/denominator, so the inner op is a single
// v_mul + global_atomic_add_f32 per feature.
__global__ void k_edge_scatter256(const float* __restrict__ xl, const int* __restrict__ ei,
                                  int E, const float* __restrict__ alpha,
                                  const float* __restrict__ inv, float* __restrict__ acc) {
  int e = blockIdx.x;
  int f = threadIdx.x;             // 0..255, head = f>>5 (== wave id: same-address
  int s, d; edge_nodes(ei, e, E, s, d);  //  alpha/inv loads broadcast per wave)
  int h = f >> 5;
  float w = alpha[e * 8 + h] * inv[d * 8 + h];
  atomicAdd(&acc[(size_t)d * 256 + f], xl[(size_t)s * 256 + f] * w);
}

// Pass 3 (layer 2, H=1,F=16): 16 threads per edge.
__global__ void k_edge_scatter16(const float* __restrict__ xl, const int* __restrict__ ei,
                                 int E, int ET, const float* __restrict__ alpha,
                                 const float* __restrict__ inv, float* __restrict__ acc) {
  int t = blockIdx.x * blockDim.x + threadIdx.x;
  int e = t >> 4, f = t & 15;
  if (e >= ET) return;
  int s, d; edge_nodes(ei, e, E, s, d);
  float w = alpha[e] * inv[d];
  atomicAdd(&acc[(size_t)d * 16 + f], xl[(size_t)s * 16 + f] * w);
}

// Between layers: h = elu(acc + b1), emitted directly as f16 for the layer-2 WMMA GEMM.
__global__ void k_elu_bias_f16(const float* __restrict__ acc, const float* __restrict__ b,
                               _Float16* __restrict__ h16, int n) {
  int i = blockIdx.x * blockDim.x + threadIdx.x;
  if (i >= n) return;
  float v = acc[i] + b[i & 255];
  v = v > 0.0f ? v : (__expf(v) - 1.0f);
  h16[i] = (_Float16)v;
}

__global__ void k_final(const float* __restrict__ acc, const float* __restrict__ b,
                        float* __restrict__ out, int n) {
  int i = blockIdx.x * blockDim.x + threadIdx.x;
  if (i >= n) return;
  float v = acc[i] + b[i & 15];
  out[i] = v > 0.0f ? v : NEG_OUT * v;
}

size_t alignUp(size_t v, size_t a) { return (v + a - 1) & ~(a - 1); }

} // namespace

extern "C" void kernel_launch(void* const* d_in, const int* in_sizes, int n_in,
                              void* d_out, int out_size, void* d_ws, size_t ws_size,
                              hipStream_t stream) {
  (void)n_in; (void)out_size; (void)ws_size;
  const float* x    = (const float*)d_in[0];
  const int*   ei   = (const int*)  d_in[1];
  const float* W1l  = (const float*)d_in[2];
  const float* W1r  = (const float*)d_in[3];
  const float* att1 = (const float*)d_in[4];
  const float* b1   = (const float*)d_in[5];
  const float* W2l  = (const float*)d_in[6];
  const float* W2r  = (const float*)d_in[7];
  const float* att2 = (const float*)d_in[8];
  const float* b2   = (const float*)d_in[9];
  float* out = (float*)d_out;

  const int Nn = in_sizes[0] / 128;   // nodes
  const int E  = in_sizes[1] / 2;     // raw edges
  const int ET = E + Nn;              // + self loops

  // ---- workspace bump allocator (256B aligned partitions) ----
  char* wsb = (char*)d_ws;
  size_t off = 0;
  auto alloc = [&](size_t bytes) -> void* {
    off = alignUp(off, 256);
    void* p = wsb + off;
    off += bytes;
    return p;
  };
  _Float16* x16   = (_Float16*)alloc((size_t)Nn * 128 * 2);
  _Float16* w1lt  = (_Float16*)alloc((size_t)256 * 128 * 2);
  _Float16* w1rt  = (_Float16*)alloc((size_t)256 * 128 * 2);
  _Float16* w2lt  = (_Float16*)alloc((size_t)16 * 256 * 2);
  _Float16* w2rt  = (_Float16*)alloc((size_t)16 * 256 * 2);
  float* xl1      = (float*)alloc((size_t)Nn * 256 * 4);
  float* xr1      = (float*)alloc((size_t)Nn * 256 * 4);   // reused as acc1 after pass 1/2
  float* alpha1   = (float*)alloc((size_t)ET * 8 * 4);     // region reused for all layer-2 arrays
  float* m1       = (float*)alloc((size_t)Nn * 8 * 4);
  float* den1     = (float*)alloc((size_t)Nn * 8 * 4);
  _Float16* h16   = (_Float16*)alloc((size_t)Nn * 256 * 2);

  float* acc1 = xr1;  // xr1 is dead after the score pass; reuse for aggregation

  // layer-2 arrays aliased into the (dead-by-then) alpha1 region (needs ~12.8MB < 22MB)
  char* l2 = (char*)alpha1;
  size_t o2 = 0;
  auto alloc2 = [&](size_t bytes) -> void* {
    o2 = alignUp(o2, 256);
    void* p = l2 + o2;
    o2 += bytes;
    return p;
  };
  float* xl2    = (float*)alloc2((size_t)Nn * 16 * 4);
  float* xr2    = (float*)alloc2((size_t)Nn * 16 * 4);
  float* alpha2 = (float*)alloc2((size_t)ET * 4);
  float* m2     = (float*)alloc2((size_t)Nn * 4);
  float* den2   = (float*)alloc2((size_t)Nn * 4);
  float* acc2   = (float*)alloc2((size_t)Nn * 16 * 4);

  dim3 b256(256);
  auto g = [](long long n, int bs) { return dim3((unsigned int)((n + bs - 1) / bs)); };

  // ---- prep: f16 conversions + weight transposes ----
  k_f32_to_f16<<<g((long long)Nn * 128, 256), b256, 0, stream>>>(x, x16, Nn * 128);
  k_transpose_f16<<<g(128 * 256, 256), b256, 0, stream>>>(W1l, w1lt, 128, 256);
  k_transpose_f16<<<g(128 * 256, 256), b256, 0, stream>>>(W1r, w1rt, 128, 256);
  k_transpose_f16<<<g(256 * 16, 256), b256, 0, stream>>>(W2l, w2lt, 256, 16);
  k_transpose_f16<<<g(256 * 16, 256), b256, 0, stream>>>(W2r, w2rt, 256, 16);
  k_fill_f32<<<g((long long)Nn * 8, 256), b256, 0, stream>>>(m1, FMIN, Nn * 8);
  k_fill_f32<<<g((long long)Nn * 8, 256), b256, 0, stream>>>(den1, 0.0f, Nn * 8);

  // ---- layer 1: WMMA GEMMs xl = x@W1l, xr = x@W1r ----
  {
    long long tiles = (long long)((Nn + 15) / 16) * (256 / 16);
    k_gemm_wmma<<<g(tiles, 4), dim3(128), 0, stream>>>(x16, w1lt, xl1, Nn, 256, 128);
    k_gemm_wmma<<<g(tiles, 4), dim3(128), 0, stream>>>(x16, w1rt, xr1, Nn, 256, 128);
  }

  // ---- layer 1: edge softmax + aggregation ----
  k_edge_score<8, 32><<<g((long long)ET * 8, 256), b256, 0, stream>>>(
      xl1, xr1, att1, ei, E, ET, alpha1, m1);
  k_edge_expsum<8><<<g((long long)ET * 8, 256), b256, 0, stream>>>(
      ei, E, ET, alpha1, m1, den1);
  k_recip<<<g((long long)Nn * 8, 256), b256, 0, stream>>>(den1, Nn * 8);
  k_fill_f32<<<g((long long)Nn * 256, 256), b256, 0, stream>>>(acc1, 0.0f, Nn * 256);
  k_edge_scatter256<<<dim3((unsigned)ET), b256, 0, stream>>>(
      xl1, ei, E, alpha1, den1, acc1);
  k_elu_bias_f16<<<g((long long)Nn * 256, 256), b256, 0, stream>>>(
      acc1, b1, h16, Nn * 256);

  // ---- layer 2: WMMA GEMMs (N=16 -> 1 tile column) ----
  {
    long long tiles = (long long)((Nn + 15) / 16);
    k_gemm_wmma<<<g(tiles, 4), dim3(128), 0, stream>>>(h16, w2lt, xl2, Nn, 16, 256);
    k_gemm_wmma<<<g(tiles, 4), dim3(128), 0, stream>>>(h16, w2rt, xr2, Nn, 16, 256);
  }
  k_fill_f32<<<g(Nn, 256), b256, 0, stream>>>(m2, FMIN, Nn);
  k_fill_f32<<<g(Nn, 256), b256, 0, stream>>>(den2, 0.0f, Nn);
  k_fill_f32<<<g((long long)Nn * 16, 256), b256, 0, stream>>>(acc2, 0.0f, Nn * 16);

  // ---- layer 2: edge softmax + aggregation ----
  k_edge_score<1, 16><<<g(ET, 256), b256, 0, stream>>>(
      xl2, xr2, att2, ei, E, ET, alpha2, m2);
  k_edge_expsum<1><<<g(ET, 256), b256, 0, stream>>>(
      ei, E, ET, alpha2, m2, den2);
  k_recip<<<g(Nn, 256), b256, 0, stream>>>(den2, Nn);
  k_edge_scatter16<<<g((long long)ET * 16, 256), b256, 0, stream>>>(
      xl2, ei, E, ET, alpha2, den2, acc2);

  // ---- output: leaky_relu(acc2 + b2, 0.01) ----
  k_final<<<g((long long)Nn * 16, 256), b256, 0, stream>>>(acc2, b2, out, Nn * 16);
}
